// MemTransformerLM_17386027614279
// MI455X (gfx1250) — compile-verified, matrix operand order checked by reading, not run on hardware
//
#include <hip/hip_runtime.h>

// ---------------------------------------------------------------------------
// Problem constants (from reference)
// ---------------------------------------------------------------------------
#define QLEN 1024
#define MLEN 1024
#define KLEN 2048   // QLEN + MLEN
#define BSZ  2
#define DMODEL 1024
#define NH   16
#define DH   64
#define DI   4096
#define NEGINF -1e30f

typedef __attribute__((ext_vector_type(16))) _Float16 v16h;
typedef __attribute__((ext_vector_type(8)))  _Float16 v8h;
typedef __attribute__((ext_vector_type(8)))  float    v8f;

// ---------------------------------------------------------------------------
// CDNA5 async copy helpers (ASYNCcnt-tracked global->LDS DMA, ISA ch10/15.18)
// ---------------------------------------------------------------------------
__device__ __forceinline__ unsigned lds_off(const void* p) {
    // generic LDS address: addr[31:0] is the LDS byte offset (ISA 10.2 aperture)
    return (unsigned)(unsigned long long)(uintptr_t)p;
}
__device__ __forceinline__ void async_b128(unsigned lds, unsigned long long gaddr) {
    asm volatile("global_load_async_to_lds_b128 %0, %1, off"
                 :: "v"(lds), "v"(gaddr) : "memory");
}
__device__ __forceinline__ void wait_async_0() { asm volatile("s_wait_asynccnt 0x0" ::: "memory"); }
__device__ __forceinline__ void wait_async_2() { asm volatile("s_wait_asynccnt 0x2" ::: "memory"); }
__device__ __forceinline__ void wait_async_4() { asm volatile("s_wait_asynccnt 0x4" ::: "memory"); }

// ---------------------------------------------------------------------------
// WMMA GEMM, all-f16 operands:  C[z] = alpha * A[z] x op(B[z%modB]) (+bias)(relu)
//   A: f16 [M][K] (lda), batched by strideA (elements)
//   OPB==1 -> B f16 [N][K] (A x B^T), async-staged, BNT=128
//   OPB==0 -> B f16 [K][N] (A x B),  sync transpose-staged, BNT=64
//   C: f32 or f16 (storeHalf) [M][N] (ldc)
// 256 threads = 8 wave32s; tile BM=128 x BNT, BK=32; double-buffered LDS,
// software-pipelined via GLOBAL_LOAD_ASYNC_TO_LDS_B128 + s_wait_asynccnt.
// Wave (wm,wn) computes 32 x (BNT/2) patch = 2 x (BNT/32) WMMA frags.
// ---------------------------------------------------------------------------
#define BM 128
#define BK 32
#define SA 40   // LDS row stride (halves): 80B rows keep 16B chunk alignment
#define SB 40

template<int OPB, int BNT>
__global__ __launch_bounds__(256)
void wmma_gemm_f16(const _Float16* __restrict__ A, const _Float16* __restrict__ B,
                   const float* __restrict__ bias, void* __restrict__ Cv,
                   int M, int N, int K, int lda, int ldb, int ldc,
                   long long strideA, long long strideB, long long strideC,
                   int modB, int relu, int storeHalf, float alpha)
{
    constexpr int NJ = BNT / 32;            // B frags per wave (N direction)
    __shared__ _Float16 lA[2][BM * SA];     // 2 x 10240 B
    __shared__ _Float16 lB[2][BNT * SB];    // 2 x (BNT*80) B

    const int tid  = threadIdx.x;
    const int lane = tid & 31;
    const int wid  = tid >> 5;
    const int wm   = wid & 3;               // 0..3 -> 32-row sub-tile
    const int wn   = wid >> 2;              // 0..1 -> (BNT/2)-col sub-tile
    const int m0   = blockIdx.y * BM;
    const int n0   = blockIdx.x * BNT;
    const int z    = blockIdx.z;

    A += (long long)z * strideA;
    B += (long long)(z % modB) * strideB;

    v8f acc[2][NJ] = {};

    // async copy geometry: each thread moves 16B chunks
    const int crow = tid >> 2;              // 0..63
    const int cchk = (tid & 3) << 3;        // half offset 0,8,16,24
    // OPB==0 manual staging geometry
    const int bkr  = tid >> 3;              // 0..31 (k row)
    const int bc8  = (tid & 7) << 3;        // 0..56 (n offset, 8 halves)

    const int fr = lane & 15;               // fragment row/col within 16
    const int kg = lane >> 4;               // K-group select (ISA 16-bit layouts)

    auto issueTile = [&](int buf, int k0) {
        #pragma unroll
        for (int p = 0; p < 2; ++p) {
            const int r = crow + p * 64;
            async_b128(lds_off(&lA[buf][r * SA + cchk]),
                       (unsigned long long)(uintptr_t)(A + (long long)(m0 + r) * lda + k0 + cchk));
        }
        if constexpr (OPB) {
            #pragma unroll
            for (int p = 0; p < BNT / 64; ++p) {
                const int r = crow + p * 64;
                async_b128(lds_off(&lB[buf][r * SB + cchk]),
                           (unsigned long long)(uintptr_t)(B + (long long)(n0 + r) * ldb + k0 + cchk));
            }
        }
    };
    auto stageBmanual = [&](int buf, int k0) {  // B is [K][N]: transpose into LDS
        const v8h bv = *(const v8h*)(B + (long long)(k0 + bkr) * ldb + n0 + bc8);
        #pragma unroll
        for (int i = 0; i < 8; ++i) lB[buf][(bc8 + i) * SB + bkr] = bv[i];
    };
    auto compute = [&](int buf) {
        v16h afrag[2], bfrag[NJ];
        #pragma unroll
        for (int i = 0; i < 2; ++i) {
            const int row = wm * 32 + i * 16 + fr;
            v8h lo = *(const v8h*)&lA[buf][row * SA + kg * 8];
            v8h hi = *(const v8h*)&lA[buf][row * SA + 16 + kg * 8];
            afrag[i] = __builtin_shufflevector(lo, hi,
                0,1,2,3,4,5,6,7,8,9,10,11,12,13,14,15);
        }
        #pragma unroll
        for (int j = 0; j < NJ; ++j) {
            const int col = wn * (BNT / 2) + j * 16 + fr;
            v8h lo = *(const v8h*)&lB[buf][col * SB + kg * 8];
            v8h hi = *(const v8h*)&lB[buf][col * SB + 16 + kg * 8];
            bfrag[j] = __builtin_shufflevector(lo, hi,
                0,1,2,3,4,5,6,7,8,9,10,11,12,13,14,15);
        }
        #pragma unroll
        for (int i = 0; i < 2; ++i)
            #pragma unroll
            for (int j = 0; j < NJ; ++j)
                acc[i][j] = __builtin_amdgcn_wmma_f32_16x16x32_f16(
                    false, afrag[i], false, bfrag[j],
                    (short)0, acc[i][j], false, false);
    };

    const int nT = K / BK;   // >= 2 for all calls in this layer
    issueTile(0, 0);
    if constexpr (!OPB) stageBmanual(0, 0);

    // steady state: unconditional pipelined body
    for (int t = 0; t < nT - 1; ++t) {
        const int cur = t & 1;
        issueTile(cur ^ 1, (t + 1) * BK);
        if constexpr (!OPB) stageBmanual(cur ^ 1, (t + 1) * BK);
        if constexpr (OPB) wait_async_4(); else wait_async_2();
        __syncthreads();
        compute(cur);
        __syncthreads();
    }
    // epilogue tile
    wait_async_0();
    __syncthreads();
    compute((nT - 1) & 1);

    // C frag layout per ISA 7.12.2 (VGPR e -> M = e + 8*(lane>>4))
    const int rhalf = (lane >> 4) << 3;
    #pragma unroll
    for (int i = 0; i < 2; ++i) {
        #pragma unroll
        for (int j = 0; j < NJ; ++j) {
            const int col = n0 + wn * (BNT / 2) + j * 16 + fr;
            const float bv = bias ? bias[col] : 0.0f;
            #pragma unroll
            for (int e = 0; e < 8; ++e) {
                const int row = m0 + wm * 32 + i * 16 + rhalf + e;
                float v = acc[i][j][e] * alpha + bv;
                if (relu) v = fmaxf(v, 0.0f);
                if (storeHalf)
                    ((_Float16*)Cv)[(long long)z * strideC + (long long)row * ldc + col] = (_Float16)v;
                else
                    ((float*)Cv)[(long long)z * strideC + (long long)row * ldc + col] = v;
            }
        }
    }
}

// ---------------------------------------------------------------------------
// f32 -> f16 bulk cast (n multiple of 1024)
// ---------------------------------------------------------------------------
__global__ __launch_bounds__(256)
void cast_f16_kernel(const float* __restrict__ src, _Float16* __restrict__ dst)
{
    const long long i = ((long long)blockIdx.x * 256 + threadIdx.x) * 4;
    const float4 f = *(const float4*)(src + i);
    _Float16* d = dst + i;
    d[0] = (_Float16)f.x; d[1] = (_Float16)f.y;
    d[2] = (_Float16)f.z; d[3] = (_Float16)f.w;
}

// heads(f16) [klen*b][3*H*DH] -> q(+bias) / k / v  f16 [b][n][len][dh]
__global__ __launch_bounds__(256)
void qkv_reorder_kernel(const _Float16* __restrict__ heads,
                        const float* __restrict__ rwb, const float* __restrict__ rrb,
                        _Float16* __restrict__ q_rw, _Float16* __restrict__ q_rr,
                        _Float16* __restrict__ kk, _Float16* __restrict__ vv)
{
    const long long gid = (long long)blockIdx.x * 256 + threadIdx.x;
    const int d = (int)(gid & (DH - 1));
    long long t = gid >> 6;
    const int j = (int)(t & (KLEN - 1));
    t >>= 11;
    const int n = (int)(t & (NH - 1));
    const int b = (int)(t >> 4);
    const _Float16* hrow = heads + ((long long)j * BSZ + b) * (3 * NH * DH);
    const long long kvidx = (((long long)(b * NH + n)) * KLEN + j) * DH + d;
    kk[kvidx] = hrow[NH * DH + n * DH + d];
    vv[kvidx] = hrow[2 * NH * DH + n * DH + d];
    if (j >= MLEN) {
        const int i = j - MLEN;
        const float qv = (float)hrow[n * DH + d];
        const long long qidx = (((long long)(b * NH + n)) * QLEN + i) * DH + d;
        q_rw[qidx] = (_Float16)(qv + rwb[n * DH + d]);
        q_rr[qidx] = (_Float16)(qv + rrb[n * DH + d]);
    }
}

// rkflat(f16) [klen][H*DH] -> rk_r f16 [n][klen][dh]
__global__ __launch_bounds__(256)
void rk_reorder_kernel(const _Float16* __restrict__ rkflat, _Float16* __restrict__ rk_r)
{
    const long long gid = (long long)blockIdx.x * 256 + threadIdx.x;
    const int d = (int)(gid & 63);
    const int j = (int)((gid >> 6) & (KLEN - 1));
    const int n = (int)(gid >> 17);
    rk_r[((long long)n * KLEN + j) * DH + d] =
        rkflat[(long long)j * (NH * DH) + n * DH + d];
}

// vec(f16) [b][n][q][dh] -> attnin f16 [(i*b+b)][n*dh+d]
__global__ __launch_bounds__(256)
void vec_reorder_kernel(const _Float16* __restrict__ vec, _Float16* __restrict__ attnin)
{
    const long long gid = (long long)blockIdx.x * 256 + threadIdx.x;
    const int d = (int)(gid & 63);
    const int i = (int)((gid >> 6) & (QLEN - 1));
    const int n = (int)((gid >> 16) & (NH - 1));
    const int b = (int)(gid >> 20);
    attnin[((long long)i * BSZ + b) * (NH * DH) + n * DH + d] =
        vec[(((long long)(b * NH + n)) * QLEN + i) * DH + d];
}

// ---------------------------------------------------------------------------
// score = (AC + rel_shift(BD)) * scale, mask, softmax -> prob (f16).
// rel_shift(BD)[i,j] == BD[i, j + QLEN-1-i] (in range wherever unmasked).
// AC row used as f32 scratch in-place; one block per (b*n, i).
// ---------------------------------------------------------------------------
__global__ __launch_bounds__(256)
void score_softmax_kernel(float* __restrict__ AC, const float* __restrict__ BD,
                          _Float16* __restrict__ prob, float scale)
{
    const int i  = blockIdx.x;
    const int bn = blockIdx.y;
    float*       acrow = AC + ((long long)bn * QLEN + i) * KLEN;
    const float* bdrow = BD + ((long long)bn * QLEN + i) * KLEN;
    _Float16*    prow  = prob + ((long long)bn * QLEN + i) * KLEN;
    const int tid = threadIdx.x;
    __shared__ float red[256];
    const int shift = QLEN - 1 - i;

    float lmax = -3.4e38f;
    for (int j = tid; j < KLEN; j += 256) {
        float s;
        if (j > i + MLEN) s = NEGINF;
        else              s = (acrow[j] + bdrow[j + shift]) * scale;
        acrow[j] = s;
        lmax = fmaxf(lmax, s);
    }
    red[tid] = lmax; __syncthreads();
    for (int o = 128; o > 0; o >>= 1) {
        if (tid < o) red[tid] = fmaxf(red[tid], red[tid + o]);
        __syncthreads();
    }
    const float m = red[0]; __syncthreads();

    float lsum = 0.0f;
    for (int j = tid; j < KLEN; j += 256) {
        const float e = __expf(acrow[j] - m);
        acrow[j] = e;
        lsum += e;
    }
    red[tid] = lsum; __syncthreads();
    for (int o = 128; o > 0; o >>= 1) {
        if (tid < o) red[tid] += red[tid + o];
        __syncthreads();
    }
    const float inv = 1.0f / red[0];
    for (int j = tid; j < KLEN; j += 256) prow[j] = (_Float16)(acrow[j] * inv);
}

// out = LN(X + Y)*g + b ; optional f16 copy for downstream GEMM input
__global__ __launch_bounds__(256)
void add_ln_kernel(const float* __restrict__ X, const float* __restrict__ Y,
                   const float* __restrict__ g, const float* __restrict__ bb,
                   float* __restrict__ out, _Float16* __restrict__ outh)
{
    const long long row = blockIdx.x;
    const float* x = X + row * DMODEL;
    const float* y = Y + row * DMODEL;
    float*       o = out + row * DMODEL;
    __shared__ float red[256];
    const int tid = threadIdx.x;

    float v[4]; float s = 0.0f;
    #pragma unroll
    for (int e = 0; e < 4; ++e) {
        const int c = tid + e * 256;
        v[e] = x[c] + y[c];
        s += v[e];
    }
    red[tid] = s; __syncthreads();
    for (int t = 128; t > 0; t >>= 1) { if (tid < t) red[tid] += red[tid + t]; __syncthreads(); }
    const float mean = red[0] * (1.0f / DMODEL); __syncthreads();

    s = 0.0f;
    #pragma unroll
    for (int e = 0; e < 4; ++e) { const float d = v[e] - mean; s += d * d; }
    red[tid] = s; __syncthreads();
    for (int t = 128; t > 0; t >>= 1) { if (tid < t) red[tid] += red[tid + t]; __syncthreads(); }
    const float rstd = rsqrtf(red[0] * (1.0f / DMODEL) + 1e-5f);

    #pragma unroll
    for (int e = 0; e < 4; ++e) {
        const int c = tid + e * 256;
        const float val = (v[e] - mean) * rstd * g[c] + bb[c];
        o[c] = val;
        if (outh) outh[row * DMODEL + c] = (_Float16)val;
    }
}

// ---------------------------------------------------------------------------
// Host-side orchestration
// ---------------------------------------------------------------------------
static inline void gemm(hipStream_t s, const _Float16* A, const _Float16* B,
                        const float* bias, void* C,
                        int M, int N, int K, int lda, int ldb, int ldc,
                        long long sA, long long sB, long long sC,
                        int nbatch, int modB, int opB, int relu, int storeHalf,
                        float alpha)
{
    if (opB) {
        dim3 grid(N / 128, M / BM, nbatch);
        wmma_gemm_f16<1, 128><<<grid, 256, 0, s>>>(A, B, bias, C, M, N, K,
            lda, ldb, ldc, sA, sB, sC, modB, relu, storeHalf, alpha);
    } else {
        dim3 grid(N / 64, M / BM, nbatch);
        wmma_gemm_f16<0, 64><<<grid, 256, 0, s>>>(A, B, bias, C, M, N, K,
            lda, ldb, ldc, sA, sB, sC, modB, relu, storeHalf, alpha);
    }
}

static inline void cast16(hipStream_t s, const float* src, _Float16* dst, long long n)
{
    cast_f16_kernel<<<(int)(n / 1024), 256, 0, s>>>(src, dst);
}

extern "C" void kernel_launch(void* const* d_in, const int* in_sizes, int n_in,
                              void* d_out, int out_size, void* d_ws, size_t ws_size,
                              hipStream_t stream) {
    (void)in_sizes; (void)n_in; (void)out_size; (void)ws_size;
    const float* w      = (const float*)d_in[0];
    const float* r      = (const float*)d_in[1];
    const float* mems   = (const float*)d_in[2];
    const float* W_qkv  = (const float*)d_in[3];
    const float* W_r    = (const float*)d_in[4];
    const float* W_o    = (const float*)d_in[5];
    const float* rwb    = (const float*)d_in[6];
    const float* rrb    = (const float*)d_in[7];
    const float* ln1_g  = (const float*)d_in[8];
    const float* ln1_b  = (const float*)d_in[9];
    const float* ff_W1  = (const float*)d_in[10];
    const float* ff_b1  = (const float*)d_in[11];
    const float* ff_W2  = (const float*)d_in[12];
    const float* ff_b2  = (const float*)d_in[13];
    const float* ln2_g  = (const float*)d_in[14];
    const float* ln2_b  = (const float*)d_in[15];
    float* out = (float*)d_out;

    // ---- workspace carve-out (256B aligned) ----
    char* p = (char*)d_ws;
    auto alloc = [&](long long bytes) -> char* {
        char* q = p; p += (bytes + 255) & ~255LL; return q;
    };
    const long long NROW  = (long long)QLEN * BSZ;           // 2048
    const long long HD3   = 3 * NH * DH;                     // 3072
    const long long HD    = NH * DH;                         // 1024

    _Float16* wh     = (_Float16*)alloc(NROW * DMODEL * 2);
    _Float16* memsh  = (_Float16*)alloc((long long)MLEN * BSZ * DMODEL * 2);
    _Float16* rh     = (_Float16*)alloc((long long)KLEN * DMODEL * 2);
    _Float16* Wqkvh  = (_Float16*)alloc(HD3 * DMODEL * 2);
    _Float16* Wrh    = (_Float16*)alloc(HD * DMODEL * 2);
    _Float16* Woh    = (_Float16*)alloc((long long)DMODEL * HD * 2);
    _Float16* ffW1h  = (_Float16*)alloc((long long)DI * DMODEL * 2);
    _Float16* ffW2h  = (_Float16*)alloc((long long)DMODEL * DI * 2);

    _Float16* heads  = (_Float16*)alloc((long long)KLEN * BSZ * HD3 * 2);
    _Float16* rkflat = (_Float16*)alloc((long long)KLEN * HD * 2);
    _Float16* q_rw   = (_Float16*)alloc((long long)BSZ * NH * QLEN * DH * 2);
    _Float16* q_rr   = (_Float16*)alloc((long long)BSZ * NH * QLEN * DH * 2);
    _Float16* kbuf   = (_Float16*)alloc((long long)BSZ * NH * KLEN * DH * 2);
    _Float16* vbuf   = (_Float16*)alloc((long long)BSZ * NH * KLEN * DH * 2);
    _Float16* rk_r   = (_Float16*)alloc((long long)NH * KLEN * DH * 2);
    float*    AC     = (float*)   alloc((long long)BSZ * NH * QLEN * KLEN * 4);
    float*    BD     = (float*)   alloc((long long)BSZ * NH * QLEN * KLEN * 4);
    _Float16* probh  = (_Float16*)alloc((long long)BSZ * NH * QLEN * KLEN * 2);
    _Float16* vec    = (_Float16*)alloc((long long)BSZ * NH * QLEN * DH * 2);
    _Float16* attnin = (_Float16*)alloc(NROW * HD * 2);
    float*    attnout= (float*)   alloc(NROW * DMODEL * 4);
    float*    out1   = (float*)   alloc(NROW * DMODEL * 4);
    _Float16* out1h  = (_Float16*)alloc(NROW * DMODEL * 2);
    _Float16* h1     = (_Float16*)alloc(NROW * DI * 2);
    float*    core   = (float*)   alloc(NROW * DMODEL * 4);

    const int BIG = 0x40000000;

    // 0) one-time f32 -> f16 casts of GEMM operands
    cast16(stream, w,     wh,    NROW * DMODEL);
    cast16(stream, mems,  memsh, (long long)MLEN * BSZ * DMODEL);
    cast16(stream, r,     rh,    (long long)KLEN * DMODEL);
    cast16(stream, W_qkv, Wqkvh, HD3 * DMODEL);
    cast16(stream, W_r,   Wrh,   HD * DMODEL);
    cast16(stream, W_o,   Woh,   (long long)DMODEL * HD);
    cast16(stream, ff_W1, ffW1h, (long long)DI * DMODEL);
    cast16(stream, ff_W2, ffW2h, (long long)DMODEL * DI);

    // 1) heads = cat(mems, w) @ W_qkv^T  (two contiguous sub-GEMMs, f16 out)
    gemm(stream, memsh, Wqkvh, nullptr, heads,
         MLEN * BSZ, (int)HD3, DMODEL, DMODEL, DMODEL, (int)HD3,
         0, 0, 0, 1, BIG, 1, 0, 1, 1.0f);
    gemm(stream, wh, Wqkvh, nullptr, heads + (long long)MLEN * BSZ * HD3,
         QLEN * BSZ, (int)HD3, DMODEL, DMODEL, DMODEL, (int)HD3,
         0, 0, 0, 1, BIG, 1, 0, 1, 1.0f);

    // 2) rkflat = r @ W_r^T (f16 out)
    gemm(stream, rh, Wrh, nullptr, rkflat,
         KLEN, (int)HD, DMODEL, DMODEL, DMODEL, (int)HD,
         0, 0, 0, 1, BIG, 1, 0, 1, 1.0f);

    // 3) reorder to per-(b,n) dense f16 layouts (+ q biases)
    qkv_reorder_kernel<<<(int)((long long)BSZ * NH * KLEN * DH / 256), 256, 0, stream>>>(
        heads, rwb, rrb, q_rw, q_rr, kbuf, vbuf);
    rk_reorder_kernel<<<(int)((long long)NH * KLEN * DH / 256), 256, 0, stream>>>(
        rkflat, rk_r);

    // 4) AC[b,n] = (q + r_w_bias) @ k^T   (batch 32, f32 out)
    gemm(stream, q_rw, kbuf, nullptr, AC,
         QLEN, KLEN, DH, DH, DH, KLEN,
         (long long)QLEN * DH, (long long)KLEN * DH, (long long)QLEN * KLEN,
         BSZ * NH, BIG, 1, 0, 0, 1.0f);

    // 5) BD[b,n] = (q + r_r_bias) @ rk^T  (rk shared over batch: modB = NH)
    gemm(stream, q_rr, rk_r, nullptr, BD,
         QLEN, KLEN, DH, DH, DH, KLEN,
         (long long)QLEN * DH, (long long)KLEN * DH, (long long)QLEN * KLEN,
         BSZ * NH, NH, 1, 0, 0, 1.0f);

    // 6) fused rel-shift + scale + mask + softmax -> prob (f16)
    {
        dim3 grid(QLEN, BSZ * NH);
        score_softmax_kernel<<<grid, 256, 0, stream>>>(AC, BD, probh, 0.125f);
    }

    // 7) vec[b,n] = prob @ v   (B is [K][N] -> opB=0; f16 out)
    gemm(stream, probh, vbuf, nullptr, vec,
         QLEN, DH, KLEN, KLEN, DH, DH,
         (long long)QLEN * KLEN, (long long)KLEN * DH, (long long)QLEN * DH,
         BSZ * NH, BIG, 0, 0, 1, 1.0f);

    // 8) gather heads back to [(i,b)][h*dh] (f16)
    vec_reorder_kernel<<<(int)((long long)BSZ * NH * QLEN * DH / 256), 256, 0, stream>>>(
        vec, attnin);

    // 9) attn_out = vec @ W_o^T (f32 out for residual)
    gemm(stream, attnin, Woh, nullptr, attnout,
         QLEN * BSZ, DMODEL, (int)HD, (int)HD, (int)HD, DMODEL,
         0, 0, 0, 1, BIG, 1, 0, 0, 1.0f);

    // 10) out1 = LN1(w + attn_out)  (f32 + f16 copy)
    add_ln_kernel<<<QLEN * BSZ, 256, 0, stream>>>(w, attnout, ln1_g, ln1_b, out1, out1h);

    // 11) h1 = relu(out1 @ ff_W1^T + b1) (f16 out)
    gemm(stream, out1h, ffW1h, ff_b1, h1,
         QLEN * BSZ, DI, DMODEL, DMODEL, DMODEL, DI,
         0, 0, 0, 1, BIG, 1, 1, 1, 1.0f);

    // 12) core = h1 @ ff_W2^T + b2 (f32 out)
    gemm(stream, h1, ffW2h, ff_b2, core,
         QLEN * BSZ, DMODEL, DI, DI, DI, DMODEL,
         0, 0, 0, 1, BIG, 1, 0, 0, 1.0f);

    // 13) d_out = LN2(out1 + core)
    add_ln_kernel<<<QLEN * BSZ, 256, 0, stream>>>(out1, core, ln2_g, ln2_b, out, nullptr);
}
